// SelfAttention_72739566125349
// MI455X (gfx1250) — matrix operand order repaired and checked by reading, not compiled, not run on hardware
//
#include <hip/hip_runtime.h>
#include <hip/hip_bf16.h>
#include <stdint.h>

// ---------------------------------------------------------------------------
// Types & helpers
// ---------------------------------------------------------------------------
typedef __attribute__((ext_vector_type(16))) __bf16 bf16x16;
typedef __attribute__((ext_vector_type(8)))  float  f32x8;
typedef __attribute__((ext_vector_type(4)))  int    i32x4;

union FragAB { uint4 u[2]; bf16x16 v; };

#define WMMA_BF16(A, B, C) \
  __builtin_amdgcn_wmma_f32_16x16x32_bf16(false, (A), false, (B), (short)0, (C), false, false)

__device__ __forceinline__ uint16_t f2bf(float f) {
  __bf16 h = (__bf16)f;
  return __builtin_bit_cast(uint16_t, h);
}

// Async global->LDS 16-byte copy (CDNA5 GLOBAL_LOAD_ASYNC_TO_LDS_B128,
// tracked by ASYNCcnt). Builtin signature: (v4i addrspace(1)*, v4i addrspace(3)*,
// imm offset, imm cpol). Falls back to a synchronous copy if unavailable.
__device__ __forceinline__ void async_copy_b128(const uint16_t* g, uint16_t* l) {
#if defined(__AMDGCN__) && __has_builtin(__builtin_amdgcn_global_load_async_to_lds_b128)
  __builtin_amdgcn_global_load_async_to_lds_b128(
      (__attribute__((address_space(1))) i32x4*)(g),
      (__attribute__((address_space(3))) i32x4*)(l), 0, 0);
#else
  *reinterpret_cast<uint4*>(l) = *reinterpret_cast<const uint4*>(g);
#endif
}

__device__ __forceinline__ void wait_async0() {
#if defined(__AMDGCN__) && __has_builtin(__builtin_amdgcn_s_wait_asynccnt)
  __builtin_amdgcn_s_wait_asynccnt(0);
#elif defined(__AMDGCN__)
  asm volatile("s_wait_asynccnt 0" ::: "memory");
#endif
}

// ---------------------------------------------------------------------------
// Kernel 0: f32 -> bf16 bulk convert (vectorized x4)
// ---------------------------------------------------------------------------
__global__ __launch_bounds__(256) void f32_to_bf16_kernel(
    const float* __restrict__ in, uint16_t* __restrict__ out, int n4) {
  int i = blockIdx.x * blockDim.x + threadIdx.x;
  if (i < n4) {
    float4 f = reinterpret_cast<const float4*>(in)[i];
    ushort4 r;
    r.x = f2bf(f.x); r.y = f2bf(f.y); r.z = f2bf(f.z); r.w = f2bf(f.w);
    reinterpret_cast<ushort4*>(out)[i] = r;
  }
}

// ---------------------------------------------------------------------------
// Generic bf16 WMMA GEMM: C = A[M,K] * W[K,N] + bias, block tile 128x128x32,
// double-buffered LDS with async global->LDS staging of the A tile.
// 8 waves: wave (wm in 0..1, wn in 0..3) owns 64x32 -> 4x2 16x16 accum tiles.
// MODE 0: qk de-interleave -> q,k bf16 [b,h,s,64]
// MODE 1: v scatter       -> v   bf16 [b,h,s,64]
// MODE 2: plain f32 out   -> [M,N]
// All dims fixed multiples of tile sizes -> no bounds checks.
// ---------------------------------------------------------------------------
template <int MODE>
__global__ __launch_bounds__(256) void gemm_bf16_kernel(
    const uint16_t* __restrict__ A,   // [M][K] bf16
    const uint16_t* __restrict__ W,   // [K][N] bf16
    const float* __restrict__ bias,   // [N]
    void* __restrict__ out0, void* __restrict__ out1,
    int M, int N, int K) {
  __shared__ uint16_t lA[2][128 * 32];  // [m][k]
  __shared__ uint16_t lB[2][128 * 32];  // [n][k]  (transposed tile of W)

  const int tid = threadIdx.x;
  const int lane = tid & 31, w = tid >> 5;
  const int wm = w >> 2, wn = w & 3;
  const int row0 = blockIdx.y * 128;
  const int col0 = blockIdx.x * 128;
  const int sel = lane >> 4, l15 = lane & 15;

  f32x8 acc[4][2];
#pragma unroll
  for (int mt = 0; mt < 4; mt++)
#pragma unroll
    for (int nt = 0; nt < 2; nt++)
#pragma unroll
      for (int r = 0; r < 8; r++) acc[mt][nt][r] = 0.f;

  const int la_row = tid >> 1, la_half = tid & 1;  // A: 128 rows x 2 half-rows
  const int lb_k = tid >> 3, lb_nseg = tid & 7;    // W: 32 k-rows x 8 n-segments

  auto stage = [&](int k0, int bufsel) {
    // A tile: straight copy -> async to LDS
    const uint16_t* gA = A + (size_t)(row0 + la_row) * K + k0 + la_half * 16;
    uint16_t* dA = &lA[bufsel][la_row * 32 + la_half * 16];
    async_copy_b128(gA, dA);
    async_copy_b128(gA + 8, dA + 8);
    // W tile: transpose on store (lane repack, synchronous)
    const uint16_t* gW = W + (size_t)(k0 + lb_k) * N + col0 + lb_nseg * 16;
    uint4 b0 = reinterpret_cast<const uint4*>(gW)[0];
    uint4 b1 = reinterpret_cast<const uint4*>(gW)[1];
    const uint16_t* e = reinterpret_cast<const uint16_t*>(&b0);
#pragma unroll
    for (int i = 0; i < 8; i++) lB[bufsel][(lb_nseg * 16 + i) * 32 + lb_k] = e[i];
    e = reinterpret_cast<const uint16_t*>(&b1);
#pragma unroll
    for (int i = 0; i < 8; i++) lB[bufsel][(lb_nseg * 16 + 8 + i) * 32 + lb_k] = e[i];
  };

  const int nIter = K >> 5;
  stage(0, 0);
  wait_async0();
  __syncthreads();

  for (int it = 0; it < nIter; ++it) {
    const int buf = it & 1;
    if (it + 1 < nIter) stage((it + 1) << 5, buf ^ 1);  // overlap next tile

    // ---- fragments from LDS (ISA 7.12.2 layouts) ----
    FragAB af[4], bfr[2];
#pragma unroll
    for (int mt = 0; mt < 4; mt++) {
      int m = wm * 64 + mt * 16 + l15;
      af[mt].u[0] = *reinterpret_cast<const uint4*>(&lA[buf][m * 32 + sel * 8]);
      af[mt].u[1] = *reinterpret_cast<const uint4*>(&lA[buf][m * 32 + 16 + sel * 8]);
    }
#pragma unroll
    for (int nt = 0; nt < 2; nt++) {
      int n = wn * 32 + nt * 16 + l15;
      bfr[nt].u[0] = *reinterpret_cast<const uint4*>(&lB[buf][n * 32 + sel * 8]);
      bfr[nt].u[1] = *reinterpret_cast<const uint4*>(&lB[buf][n * 32 + 16 + sel * 8]);
    }
#pragma unroll
    for (int mt = 0; mt < 4; mt++)
#pragma unroll
      for (int nt = 0; nt < 2; nt++)
        acc[mt][nt] = WMMA_BF16(af[mt].v, bfr[nt].v, acc[mt][nt]);

    wait_async0();     // next tile's async copies landed in LDS
    __syncthreads();   // single barrier per K-step
  }

  // ---- epilogue ----
#pragma unroll
  for (int mt = 0; mt < 4; mt++) {
#pragma unroll
    for (int nt = 0; nt < 2; nt++) {
#pragma unroll
      for (int r = 0; r < 8; r++) {
        int gr = row0 + wm * 64 + mt * 16 + r + sel * 8;
        int gc = col0 + wn * 32 + nt * 16 + l15;
        float val = acc[mt][nt][r] + bias[gc];
        if (MODE == 2) {
          reinterpret_cast<float*>(out0)[(size_t)gr * N + gc] = val;
        } else if (MODE == 0) {
          int bz = gr >> 12, s = gr & 4095;
          int h = gc >> 7, rem = gc & 127, d = rem >> 1;
          uint16_t* dst = (rem & 1) ? (uint16_t*)out1 : (uint16_t*)out0;
          dst[((size_t)((bz * 8 + h) * 4096 + s)) * 64 + d] = f2bf(val);
        } else {  // MODE 1
          int bz = gr >> 12, s = gr & 4095;
          int h = gc >> 6, d = gc & 63;
          ((uint16_t*)out0)[((size_t)((bz * 8 + h) * 4096 + s)) * 64 + d] = f2bf(val);
        }
      }
    }
  }
}

// ---------------------------------------------------------------------------
// Fused flash-attention: block = (b, h, 128 query rows), 8 waves of 16 rows.
// Double-buffered 64-row K/V tiles (K async to LDS, V transposed on store),
// online softmax in registers, P re-fragmented through wave-private LDS.
// q,k,v bf16 [b,h,s,64]. Writes o bf16 [b, s, h*64+d].
// ---------------------------------------------------------------------------
__global__ __launch_bounds__(256) void attention_kernel(
    const uint16_t* __restrict__ q, const uint16_t* __restrict__ k,
    const uint16_t* __restrict__ v, uint16_t* __restrict__ o) {
  __shared__ uint16_t lK[2][64 * 64];   // [j][d]
  __shared__ uint16_t lVt[2][64 * 64];  // [d][j]  (transposed V tile)
  __shared__ uint16_t lP[8 * 16 * 64];  // per-wave 16x64 P tile

  const int tid = threadIdx.x, lane = tid & 31, w = tid >> 5;
  const int sel = lane >> 4, l15 = lane & 15;
  const int ib = blockIdx.x;   // 0..31 query blocks
  const int h = blockIdx.y;    // 0..7
  const int bz = blockIdx.z;   // 0..1
  const size_t head = ((size_t)(bz * 8 + h)) * 4096 * 64;
  const uint16_t* qg = q + head;
  const uint16_t* kg = k + head;
  const uint16_t* vg = v + head;
  const int i0 = ib * 128 + w * 16;

  // Q A-fragments (16 rows x 64 K) held in registers for the whole pass
  FragAB qf[2];
  {
    size_t m = (size_t)(i0 + l15) * 64;
    qf[0].u[0] = *reinterpret_cast<const uint4*>(qg + m + sel * 8);
    qf[0].u[1] = *reinterpret_cast<const uint4*>(qg + m + 16 + sel * 8);
    qf[1].u[0] = *reinterpret_cast<const uint4*>(qg + m + 32 + sel * 8);
    qf[1].u[1] = *reinterpret_cast<const uint4*>(qg + m + 48 + sel * 8);
  }

  f32x8 acc[4];
#pragma unroll
  for (int dt = 0; dt < 4; dt++)
#pragma unroll
    for (int r = 0; r < 8; r++) acc[dt][r] = 0.f;
  float mi[8], li[8];
#pragma unroll
  for (int r = 0; r < 8; r++) { mi[r] = -1e30f; li[r] = 0.f; }

  const int ld_row = tid >> 2, ld_seg = tid & 3;
  uint16_t* lPw = lP + w * 1024;

  auto stageKV = [&](int jb, int bufsel) {
    // K tile: straight copy -> async to LDS
    const uint16_t* gk = kg + (size_t)(jb + ld_row) * 64 + ld_seg * 16;
    uint16_t* dk = &lK[bufsel][ld_row * 64 + ld_seg * 16];
    async_copy_b128(gk, dk);
    async_copy_b128(gk + 8, dk + 8);
    // V tile: transpose on store
    const uint4* gv = reinterpret_cast<const uint4*>(
        vg + (size_t)(jb + ld_row) * 64 + ld_seg * 16);
    uint4 v0 = gv[0], v1 = gv[1];
    const uint16_t* e = reinterpret_cast<const uint16_t*>(&v0);
#pragma unroll
    for (int i = 0; i < 8; i++) lVt[bufsel][(ld_seg * 16 + i) * 64 + ld_row] = e[i];
    e = reinterpret_cast<const uint16_t*>(&v1);
#pragma unroll
    for (int i = 0; i < 8; i++) lVt[bufsel][(ld_seg * 16 + 8 + i) * 64 + ld_row] = e[i];
  };

  stageKV(0, 0);
  wait_async0();
  __syncthreads();

  for (int it = 0; it < 64; ++it) {  // 4096 / 64 key blocks
    const int buf = it & 1;
    if (it + 1 < 64) stageKV((it + 1) * 64, buf ^ 1);  // overlap next K/V tile

    // ---- S = scale * Q K^T  (4 score tiles of 16x16, 2 WMMAs each) ----
    f32x8 sv[4];
#pragma unroll
    for (int nt = 0; nt < 4; nt++) {
      FragAB kb0, kb1;
      int n = nt * 16 + l15;
      kb0.u[0] = *reinterpret_cast<const uint4*>(&lK[buf][n * 64 + sel * 8]);
      kb0.u[1] = *reinterpret_cast<const uint4*>(&lK[buf][n * 64 + 16 + sel * 8]);
      kb1.u[0] = *reinterpret_cast<const uint4*>(&lK[buf][n * 64 + 32 + sel * 8]);
      kb1.u[1] = *reinterpret_cast<const uint4*>(&lK[buf][n * 64 + 48 + sel * 8]);
      f32x8 t;
#pragma unroll
      for (int r = 0; r < 8; r++) t[r] = 0.f;
      t = WMMA_BF16(qf[0].v, kb0.v, t);
      t = WMMA_BF16(qf[1].v, kb1.v, t);
      sv[nt] = t * 0.125f;  // D_HEAD^-0.5
    }

    // ---- online softmax (row-wise over 64 keys; rows striped across lanes) --
#pragma unroll
    for (int r = 0; r < 8; r++) {
      float tm = fmaxf(fmaxf(sv[0][r], sv[1][r]), fmaxf(sv[2][r], sv[3][r]));
#pragma unroll
      for (int off = 8; off >= 1; off >>= 1) tm = fmaxf(tm, __shfl_xor(tm, off, 32));
      float mn = fmaxf(mi[r], tm);
      float al = __expf(mi[r] - mn);
      mi[r] = mn;
      float rs = 0.f;
#pragma unroll
      for (int nt = 0; nt < 4; nt++) {
        float p = __expf(sv[nt][r] - mn);
        rs += p;
        lPw[(r + sel * 8) * 64 + nt * 16 + l15] = f2bf(p);
      }
#pragma unroll
      for (int off = 8; off >= 1; off >>= 1) rs += __shfl_xor(rs, off, 32);
      li[r] = li[r] * al + rs;
#pragma unroll
      for (int dt = 0; dt < 4; dt++) acc[dt][r] *= al;
    }

    // wave-private LDS round trip: only need this wave's DS ops complete
    asm volatile("s_wait_dscnt 0" ::: "memory");

    // ---- O += P V : P A-frags from LDS, V^T B-frags ----
    FragAB pf[2];
    pf[0].u[0] = *reinterpret_cast<const uint4*>(&lPw[l15 * 64 + sel * 8]);
    pf[0].u[1] = *reinterpret_cast<const uint4*>(&lPw[l15 * 64 + 16 + sel * 8]);
    pf[1].u[0] = *reinterpret_cast<const uint4*>(&lPw[l15 * 64 + 32 + sel * 8]);
    pf[1].u[1] = *reinterpret_cast<const uint4*>(&lPw[l15 * 64 + 48 + sel * 8]);
#pragma unroll
    for (int dt = 0; dt < 4; dt++) {
      FragAB vb0, vb1;
      int n = dt * 16 + l15;
      vb0.u[0] = *reinterpret_cast<const uint4*>(&lVt[buf][n * 64 + sel * 8]);
      vb0.u[1] = *reinterpret_cast<const uint4*>(&lVt[buf][n * 64 + 16 + sel * 8]);
      vb1.u[0] = *reinterpret_cast<const uint4*>(&lVt[buf][n * 64 + 32 + sel * 8]);
      vb1.u[1] = *reinterpret_cast<const uint4*>(&lVt[buf][n * 64 + 48 + sel * 8]);
      acc[dt] = WMMA_BF16(pf[0].v, vb0.v, acc[dt]);
      acc[dt] = WMMA_BF16(pf[1].v, vb1.v, acc[dt]);
    }

    wait_async0();     // next K tile's async copies landed
    __syncthreads();   // single barrier per key block
  }

  // ---- normalize and write o = [b, s, (h d)] as bf16 ----
#pragma unroll
  for (int dt = 0; dt < 4; dt++) {
#pragma unroll
    for (int r = 0; r < 8; r++) {
      int i = i0 + r + sel * 8;
      int col = h * 64 + dt * 16 + l15;
      float val = acc[dt][r] / li[r];
      o[((size_t)(bz * 4096 + i)) * 512 + col] = f2bf(val);
    }
  }
}

// ---------------------------------------------------------------------------
// LayerNorm + residual + 1/sqrt(2). One block per row of 1024.
// ---------------------------------------------------------------------------
__global__ __launch_bounds__(256) void ln_residual_kernel(
    const float* __restrict__ ypre, const float* __restrict__ x,
    const float* __restrict__ gamma, const float* __restrict__ beta,
    float* __restrict__ out) {
  __shared__ float s1[256], s2[256];
  const int row = blockIdx.x, t = threadIdx.x;
  const float* yp = ypre + (size_t)row * 1024;
  const float* xr = x + (size_t)row * 1024;
  float* op = out + (size_t)row * 1024;

  float4 y = reinterpret_cast<const float4*>(yp)[t];
  s1[t] = y.x + y.y + y.z + y.w;
  s2[t] = y.x * y.x + y.y * y.y + y.z * y.z + y.w * y.w;
  __syncthreads();
  for (int stride = 128; stride > 0; stride >>= 1) {
    if (t < stride) { s1[t] += s1[t + stride]; s2[t] += s2[t + stride]; }
    __syncthreads();
  }
  const float mu = s1[0] * (1.0f / 1024.0f);
  const float var = s2[0] * (1.0f / 1024.0f) - mu * mu;
  const float rstd = rsqrtf(var + 1e-5f);

  float4 g = reinterpret_cast<const float4*>(gamma)[t];
  float4 be = reinterpret_cast<const float4*>(beta)[t];
  float4 xv = reinterpret_cast<const float4*>(xr)[t];
  const float RS2 = 0.70710678118654752440f;
  float4 r;
  r.x = ((y.x - mu) * rstd * g.x + be.x + xv.x) * RS2;
  r.y = ((y.y - mu) * rstd * g.y + be.y + xv.y) * RS2;
  r.z = ((y.z - mu) * rstd * g.z + be.z + xv.z) * RS2;
  r.w = ((y.w - mu) * rstd * g.w + be.w + xv.w) * RS2;
  reinterpret_cast<float4*>(op)[t] = r;
}

// ---------------------------------------------------------------------------
// Host entry
// ---------------------------------------------------------------------------
extern "C" void kernel_launch(void* const* d_in, const int* in_sizes, int n_in,
                              void* d_out, int out_size, void* d_ws, size_t ws_size,
                              hipStream_t stream) {
  (void)in_sizes; (void)n_in; (void)out_size; (void)ws_size;
  const float* x     = (const float*)d_in[0];
  const float* W_qk  = (const float*)d_in[1];
  const float* b_qk  = (const float*)d_in[2];
  const float* W_v   = (const float*)d_in[3];
  const float* b_v   = (const float*)d_in[4];
  const float* W_out = (const float*)d_in[5];
  const float* b_out = (const float*)d_in[6];
  const float* ln_g  = (const float*)d_in[7];
  const float* ln_b  = (const float*)d_in[8];
  float* out = (float*)d_out;

  char* ws = (char*)d_ws;
  size_t off = 0;
  auto alloc = [&](size_t bytes) {
    void* p = ws + off;
    off += (bytes + 255) & ~(size_t)255;
    return p;
  };
  uint16_t* xbf = (uint16_t*)alloc(8388608ull * 2);  // [8192,1024]
  uint16_t* wqk = (uint16_t*)alloc(1048576ull * 2);  // [1024,1024]
  uint16_t* wv  = (uint16_t*)alloc(524288ull * 2);   // [1024,512]
  uint16_t* wo  = (uint16_t*)alloc(524288ull * 2);   // [512,1024]
  uint16_t* qb  = (uint16_t*)alloc(4194304ull * 2);  // [2,8,4096,64]
  uint16_t* kb  = (uint16_t*)alloc(4194304ull * 2);
  uint16_t* vb  = (uint16_t*)alloc(4194304ull * 2);
  uint16_t* ob  = (uint16_t*)alloc(4194304ull * 2);  // [8192,512]
  float*   ypre = (float*)alloc(8388608ull * 4);     // [8192,1024]

  f32_to_bf16_kernel<<<8192, 256, 0, stream>>>(x, xbf, 2097152);
  f32_to_bf16_kernel<<<1024, 256, 0, stream>>>(W_qk, wqk, 262144);
  f32_to_bf16_kernel<<<512, 256, 0, stream>>>(W_v, wv, 131072);
  f32_to_bf16_kernel<<<512, 256, 0, stream>>>(W_out, wo, 131072);

  gemm_bf16_kernel<0><<<dim3(8, 64), 256, 0, stream>>>(
      xbf, wqk, b_qk, qb, kb, 8192, 1024, 1024);
  gemm_bf16_kernel<1><<<dim3(4, 64), 256, 0, stream>>>(
      xbf, wv, b_v, vb, nullptr, 8192, 512, 1024);

  attention_kernel<<<dim3(32, 8, 2), 256, 0, stream>>>(qb, kb, vb, ob);

  gemm_bf16_kernel<2><<<dim3(8, 64), 256, 0, stream>>>(
      ob, wo, b_out, ypre, nullptr, 8192, 1024, 512);

  ln_residual_kernel<<<8192, 256, 0, stream>>>(ypre, x, ln_g, ln_b, out);
}
